// ClusterGCNNet_87600152969646
// MI455X (gfx1250) — matrix operand order repaired and checked by reading, not compiled, not run on hardware
//
#include <hip/hip_runtime.h>
#include <hip/hip_bf16.h>

// ---------------------------------------------------------------------------
// GraphSAGE (3 layers, mean aggregation) for gfx1250.
//   layer: out = relu( mean_agg @ Wl + h @ Wr + b )   (no relu on last)
// Fused GEMM: A_cat = [agg_mean | h]  (N x 512, bf16),  W_cat = [Wl; Wr]
// GEMM runs on v_wmma_f32_16x16x32_bf16 (wave32, 16x16 tiles, K=32/step).
// ---------------------------------------------------------------------------

typedef __attribute__((ext_vector_type(16))) __bf16 v16bf;
typedef __attribute__((ext_vector_type(8)))  float  v8f;

union Frag32B {            // 16 bf16 = 32 bytes = one WMMA A/B operand
    v16bf v;
    uint4 u[2];
};

static __device__ __forceinline__ unsigned short f2bf(float f) {
    union { float f; unsigned u; } v;
    v.f = f;
    // round-to-nearest-even f32 -> bf16
    unsigned r = v.u + 0x7FFFu + ((v.u >> 16) & 1u);
    return (unsigned short)(r >> 16);
}

// ---------------------------------------------------------------------------
// Kernel 1: edge scatter.  One wave per edge; 32 lanes x 8 floats cover the
// 256-wide feature row.  x is 51MB -> lives in L2, so gathers + f32 atomics
// stay on-chip.
// ---------------------------------------------------------------------------
__launch_bounds__(256)
__global__ void sage_scatter(const float* __restrict__ x,
                             const long long* __restrict__ ei,
                             float* __restrict__ agg,
                             float* __restrict__ deg,
                             int E)
{
    const int t    = blockIdx.x * 256 + threadIdx.x;
    const int e    = t >> 5;
    const int lane = t & 31;
    if (e >= E) return;
    const int src = (int)ei[e];
    const int dst = (int)ei[(size_t)E + e];

    const float4* xs = (const float4*)(x + (size_t)src * 256 + lane * 8);
    const float4 a = xs[0];
    const float4 b = xs[1];

    float* d = agg + (size_t)dst * 256 + lane * 8;
    atomicAdd(d + 0, a.x); atomicAdd(d + 1, a.y);
    atomicAdd(d + 2, a.z); atomicAdd(d + 3, a.w);
    atomicAdd(d + 4, b.x); atomicAdd(d + 5, b.y);
    atomicAdd(d + 6, b.z); atomicAdd(d + 7, b.w);
    if (lane == 0) atomicAdd(deg + dst, 1.0f);
}

// ---------------------------------------------------------------------------
// Kernel 2: mean-divide + f32->bf16 cast of both GEMM A inputs (4 elems/thr).
// ---------------------------------------------------------------------------
__launch_bounds__(256)
__global__ void sage_mean_cast(const float* __restrict__ agg,
                               const float* __restrict__ deg,
                               const float* __restrict__ h,
                               unsigned short* __restrict__ aggb,
                               unsigned short* __restrict__ hb,
                               int nRows)
{
    const size_t i4     = (size_t)blockIdx.x * 256 + threadIdx.x;
    const size_t total4 = (size_t)nRows * 64;          // (nRows*256)/4
    if (i4 >= total4) return;
    const size_t base = i4 * 4;
    const int row     = (int)(base >> 8);
    const float invd  = 1.0f / fmaxf(deg[row], 1.0f);

    const float4 a  = *(const float4*)(agg + base);
    const float4 hv = *(const float4*)(h + base);
    ushort4 ua, uh;
    ua.x = f2bf(a.x * invd);  ua.y = f2bf(a.y * invd);
    ua.z = f2bf(a.z * invd);  ua.w = f2bf(a.w * invd);
    uh.x = f2bf(hv.x);        uh.y = f2bf(hv.y);
    uh.z = f2bf(hv.z);        uh.w = f2bf(hv.w);
    *(ushort4*)(aggb + base) = ua;
    *(ushort4*)(hb + base)   = uh;
}

// ---------------------------------------------------------------------------
// Kernel 3: pack W_cat = [Wl ; Wr] (f32, row-major [k][n]) into the dense
// 32x16 bf16 B-fragment layout:
//   flat = ((kt*ntiles + nt)*32 + lane)*16 + i
//   lane<16 : column n = nt*16+lane,    K = kt*32 + i (i<8), kt*32 + 16 + (i-8) (i>=8)
//   lane>=16: column n = nt*16+lane-16, K = kt*32 + 8 + i (i<8), kt*32 + 24 + (i-8) (i>=8)
// so the GEMM B load is 32 contiguous bytes per lane.
// ---------------------------------------------------------------------------
__launch_bounds__(256)
__global__ void sage_pack_w(const float* __restrict__ Wl,
                            const float* __restrict__ Wr,
                            unsigned short* __restrict__ Wp,
                            int dIn, int dOut)
{
    const int t     = blockIdx.x * 256 + threadIdx.x;
    const int total = 2 * dIn * dOut;
    if (t >= total) return;
    const int i      = t & 15;
    const int lane   = (t >> 4) & 31;
    const int tile   = t >> 9;
    const int ntiles = dOut >> 4;
    const int nt     = tile % ntiles;
    const int kt     = tile / ntiles;
    const int n      = nt * 16 + (lane & 15);
    const int k      = kt * 32 + ((lane >> 4) << 3) + (i & 7) + ((i >> 3) << 4);
    const float v = (k < dIn) ? Wl[(size_t)k * dOut + n]
                              : Wr[(size_t)(k - dIn) * dOut + n];
    Wp[t] = f2bf(v);
}

// ---------------------------------------------------------------------------
// Kernel 4: WMMA GEMM.  out[N,dOut] = [aggb|hb] @ Wp + bias (opt. ReLU)
// Block = 256 thr = 8 waves -> 128 rows/block.  Each wave: 16 rows x NT*16
// cols, K=512 walked in 16 steps of 32 (k<256 from aggb, else hb).
// A fragment (16x32 bf16): lane l -> row (l&15); elems 0..7 = K kk+(l>>4)*8,
// elems 8..15 = same +16  (ISA 7.12.2 16-bit A layout).
// ---------------------------------------------------------------------------
template<int NT>
__launch_bounds__(256)
__global__ void sage_gemm_bf16(const unsigned short* __restrict__ A0,  // aggb: K 0..255
                               const unsigned short* __restrict__ A1,  // hb:   K 256..511
                               const unsigned short* __restrict__ Wp,
                               const float* __restrict__ bias,
                               float* __restrict__ out,
                               int nRows, int dOut, int relu)
{
    const int lane = threadIdx.x & 31;
    const int wave = threadIdx.x >> 5;
    const int row0 = blockIdx.x * 128 + wave * 16;
    if (row0 >= nRows) return;                 // wave-uniform (rows %16 == 0)

    const int ntilesTotal = dOut >> 4;
    const int ntBase      = blockIdx.y * NT;
    const int aRow        = row0 + (lane & 15);
    const int kOff        = (lane >> 4) * 8;   // bf16 elems within 32-wide chunk

    v8f acc[NT] = {};

#pragma unroll
    for (int kt = 0; kt < 16; ++kt) {
        const unsigned short* Abase = (kt < 8) ? A0 : A1;
        const int kk = (kt & 7) * 32;

        Frag32B a;
        const unsigned short* pa = Abase + (size_t)aRow * 256 + kk + kOff;
        a.u[0] = *(const uint4*)(pa);          // K = kk+kOff .. +7
        a.u[1] = *(const uint4*)(pa + 16);     // K = kk+kOff+16 .. +23

        const unsigned short* pbBase =
            Wp + (((size_t)kt * ntilesTotal + ntBase) * 32 + lane) * 16;
#pragma unroll
        for (int j = 0; j < NT; ++j) {
            Frag32B b;
            const unsigned short* pb = pbBase + (size_t)j * (32 * 16);
            b.u[0] = *(const uint4*)(pb);
            b.u[1] = *(const uint4*)(pb + 8);
            acc[j] = __builtin_amdgcn_wmma_f32_16x16x32_bf16(
                         false, a.v, false, b.v, (short)0, acc[j], false, false);
        }
    }

    // C/D layout (ISA 7.12.2): VGPR r -> row r (lanes 0-15) / row r+8 (16-31),
    // col = lane & 15.
    const int rowAdd = (lane < 16) ? 0 : 8;
#pragma unroll
    for (int j = 0; j < NT; ++j) {
        const int col  = (ntBase + j) * 16 + (lane & 15);
        const float bv = bias[col];
#pragma unroll
        for (int r = 0; r < 8; ++r) {
            const int row = row0 + rowAdd + r;
            float v = acc[j][r] + bv;
            if (relu) v = fmaxf(v, 0.0f);
            out[(size_t)row * dOut + col] = v;
        }
    }
}

// ---------------------------------------------------------------------------
extern "C" void kernel_launch(void* const* d_in, const int* in_sizes, int n_in,
                              void* d_out, int out_size, void* d_ws, size_t ws_size,
                              hipStream_t stream)
{
    const float*      x  = (const float*)d_in[0];
    const long long*  ei = (const long long*)d_in[1];
    const float* Wl[3] = {(const float*)d_in[2], (const float*)d_in[5], (const float*)d_in[8]};
    const float* Wr[3] = {(const float*)d_in[3], (const float*)d_in[6], (const float*)d_in[9]};
    const float* bb[3] = {(const float*)d_in[4], (const float*)d_in[7], (const float*)d_in[10]};

    const int N = in_sizes[0] / 256;
    const int E = in_sizes[1] / 2;

    // workspace carve-out (256B aligned)
    char* ws = (char*)d_ws;
    auto carve = [&](size_t bytes) {
        char* p = ws;
        ws += (bytes + 255) & ~(size_t)255;
        return p;
    };
    float*          agg  = (float*)carve((size_t)N * 256 * 4);
    float*          deg  = (float*)carve((size_t)N * 4);
    float*          h    = (float*)carve((size_t)N * 256 * 4);   // layer activations (f32)
    unsigned short* aggb = (unsigned short*)carve((size_t)N * 256 * 2);
    unsigned short* hb   = (unsigned short*)carve((size_t)N * 256 * 2);
    unsigned short* wp   = (unsigned short*)carve((size_t)512 * 256 * 2);

    const dim3 blk(256);
    const int scatterBlocks = (E * 32 + 255) / 256;
    const int cvtBlocks     = (N * 64 + 255) / 256;
    const int gemmBlocksX   = (N + 127) / 128;

    for (int layer = 0; layer < 3; ++layer) {
        const float* hin = (layer == 0) ? x : h;
        const int dOut   = (layer == 2) ? 64 : 256;

        hipMemsetAsync(agg, 0, (size_t)N * 256 * 4, stream);
        hipMemsetAsync(deg, 0, (size_t)N * 4, stream);

        sage_scatter<<<scatterBlocks, blk, 0, stream>>>(hin, ei, agg, deg, E);
        sage_mean_cast<<<cvtBlocks, blk, 0, stream>>>(agg, deg, hin, aggb, hb, N);

        const int packTotal = 512 * dOut;
        sage_pack_w<<<(packTotal + 255) / 256, blk, 0, stream>>>(
            Wl[layer], Wr[layer], wp, 256, dOut);

        if (layer < 2) {
            // GEMM reads bf16 staging buffers, so it may safely overwrite h.
            sage_gemm_bf16<8><<<dim3(gemmBlocksX, 2), blk, 0, stream>>>(
                aggb, hb, wp, bb[layer], h, N, 256, 1);
        } else {
            sage_gemm_bf16<4><<<dim3(gemmBlocksX, 1), blk, 0, stream>>>(
                aggb, hb, wp, bb[layer], (float*)d_out, N, 64, 0);
        }
    }
}
